// LSTM_63720134803697
// MI455X (gfx1250) — matrix operand order repaired and checked by reading, not compiled, not run on hardware
//
#include <hip/hip_runtime.h>
#include <hip/hip_bf16.h>
#include <math.h>

// Problem sizes (match reference)
#define LL 512
#define BB 256
#define DD 256
#define HH 256

typedef _Float16 v16h __attribute__((ext_vector_type(16)));
typedef _Float16 v8h  __attribute__((ext_vector_type(8)));
typedef float    v8f  __attribute__((ext_vector_type(8)));

// ---------------------------------------------------------------------------
// WMMA fragment loaders (wave32, 16x16x32 f16, per CDNA5 ISA §7.12.2)
// A (16x32, MxK): lane L -> row M=L&15; lanes 0-15 hold K {0..7,16..23},
//                 lanes 16-31 hold K {8..15,24..31}.
// B (32x16, KxN): lane L -> col N=L&15; lanes 0-15 K {0..7,16..23},
//                 lanes 16-31 K {8..15,24..31}.
// C/D (16x16 f32): lane L -> col N=L&15; VGPR v -> row M = v + 8*(L>>4).
// ---------------------------------------------------------------------------
__device__ __forceinline__ v16h load_a_frag_g(const _Float16* __restrict__ base,
                                              int ld, int lid) {
  const _Float16* p = base + (size_t)(lid & 15) * ld + ((lid & 16) ? 8 : 0);
  v8h lo = *(const v8h*)(p);
  v8h hi = *(const v8h*)(p + 16);
  return __builtin_shufflevector(lo, hi, 0, 1, 2, 3, 4, 5, 6, 7,
                                 8, 9, 10, 11, 12, 13, 14, 15);
}

__device__ __forceinline__ v16h load_a_frag_lds(const _Float16* base, int ld,
                                                int lid) {
  const _Float16* p = base + (size_t)(lid & 15) * ld + ((lid & 16) ? 8 : 0);
  v8h lo = *(const v8h*)(p);           // ds_load_b128
  v8h hi = *(const v8h*)(p + 16);      // ds_load_b128
  return __builtin_shufflevector(lo, hi, 0, 1, 2, 3, 4, 5, 6, 7,
                                 8, 9, 10, 11, 12, 13, 14, 15);
}

__device__ __forceinline__ v16h load_b_frag(const _Float16* __restrict__ base,
                                            int ld, int lid) {
  const _Float16* p = base + (size_t)(lid & 15) * ld + ((lid & 16) ? 8 : 0);
  v8h lo = *(const v8h*)(p);
  v8h hi = *(const v8h*)(p + 16);
  return __builtin_shufflevector(lo, hi, 0, 1, 2, 3, 4, 5, 6, 7,
                                 8, 9, 10, 11, 12, 13, 14, 15);
}

// Async 16-byte copy global -> LDS (ASYNCcnt path, CDNA5 ISA)
__device__ __forceinline__ void async_copy_b128(unsigned lds_byte_addr,
                                                const void* gaddr) {
  asm volatile("global_load_async_to_lds_b128 %0, %1, off"
               :: "v"(lds_byte_addr), "v"(gaddr)
               : "memory");
}

__device__ __forceinline__ void wait_asynccnt0() {
  asm volatile("s_wait_asynccnt 0x0" ::: "memory");
}

__device__ __forceinline__ float sigmoidf_(float x) {
  return 1.0f / (1.0f + __expf(-x));
}

// ---------------------------------------------------------------------------
// Prep kernels
// ---------------------------------------------------------------------------
__global__ __launch_bounds__(256) void convert_weights(
    const float* __restrict__ W0, const float* __restrict__ W1,
    const float* __restrict__ W2, const float* __restrict__ W3,
    _Float16* __restrict__ Wh16, _Float16* __restrict__ Wx16) {
  int idx = blockIdx.x * 256 + threadIdx.x;      // 4 * 256 * 512 total
  if (idx >= 4 * HH * (HH + DD)) return;
  int g = idx / (HH * (HH + DD));
  int r = idx % (HH * (HH + DD));
  int h = r / (HH + DD);
  int k = r % (HH + DD);
  const float* W = (g == 0) ? W0 : (g == 1) ? W1 : (g == 2) ? W2 : W3;
  float v = W[(size_t)h * (HH + DD) + k];
  if (k < HH)
    Wh16[(size_t)g * HH * HH + (size_t)h * HH + k] = (_Float16)v;
  else
    Wx16[(size_t)g * HH * DD + (size_t)h * DD + (k - HH)] = (_Float16)v;
}

__global__ __launch_bounds__(256) void convert_x(const float* __restrict__ x,
                                                 _Float16* __restrict__ x16,
                                                 int n) {
  int i = blockIdx.x * 256 + threadIdx.x;
  if (i < n) x16[i] = (_Float16)x[i];
}

__global__ __launch_bounds__(256) void init_state(
    const float* __restrict__ h0, const float* __restrict__ c0,
    float* __restrict__ hState, float* __restrict__ cState,
    _Float16* __restrict__ h16) {
  int i = blockIdx.x * 256 + threadIdx.x;   // B*H = 65536
  if (i >= BB * HH) return;
  hState[i] = h0[i];
  cState[i] = c0[i];
  h16[i] = (_Float16)h0[i];
}

// ---------------------------------------------------------------------------
// Parallel x-contribution GEMM (M=L*B=131072, N=4H=1024, K=D=256).
// Block: 8 waves. Block stages a 64-row A panel (32 KB) in LDS via async
// copies; each wave owns one gn=(gate,ntile) and computes 4 m-subtiles,
// reusing its B fragment 4x and the shared A panel 8x.
// Output Xbuf is f16 (halves the dominant HBM stream), bias folded in.
// ---------------------------------------------------------------------------
__global__ __launch_bounds__(256) void x_gemm(
    const _Float16* __restrict__ x16, const _Float16* __restrict__ Wx16,
    const float* __restrict__ b0, const float* __restrict__ b1,
    const float* __restrict__ b2, const float* __restrict__ b3,
    _Float16* __restrict__ Xbuf16) {
  __shared__ __attribute__((aligned(16))) _Float16 Ash[64 * DD];  // 32 KB

  int tid  = threadIdx.x;
  int lid  = tid & 31;
  int wave = tid >> 5;
  int mtg  = blockIdx.x >> 3;        // 2048 m-groups of 64 rows
  int gng  = blockIdx.x & 7;         // 8 gn-groups of 8
  int gn   = gng * 8 + wave;         // 0..63
  int g    = gn >> 4;
  int n0   = (gn & 15) * 16;

  // Stage A panel: rows m = mtg*64 .. +63, 256 halves each (512 B/row).
  // 256 threads x 128 B each = 8 async b128 copies per thread.
  {
    int r = tid >> 2;                       // 0..63
    int cb = (tid & 3) * 128;               // byte offset within row
    const char* src = (const char*)(x16 + ((size_t)mtg * 64 + r) * DD) + cb;
    unsigned dst = (unsigned)(size_t)(&Ash[0]) + (unsigned)(r * 512 + cb);
#pragma unroll
    for (int j = 0; j < 8; ++j)
      async_copy_b128(dst + j * 16, src + j * 16);
  }
  wait_asynccnt0();
  __syncthreads();

  v8f zero = {0.f, 0.f, 0.f, 0.f, 0.f, 0.f, 0.f, 0.f};
  v8f acc[4] = {zero, zero, zero, zero};

  const _Float16* Bbase = Wx16 + (size_t)g * HH * DD + (size_t)n0 * DD;
#pragma unroll
  for (int k0 = 0; k0 < DD; k0 += 32) {
    v16h b = load_b_frag(Bbase + k0, DD, lid);
#pragma unroll
    for (int ms = 0; ms < 4; ++ms) {
      v16h a = load_a_frag_lds(&Ash[(ms * 16) * DD + k0], DD, lid);
      acc[ms] = __builtin_amdgcn_wmma_f32_16x16x32_f16(
          false, a, false, b, (short)0, acc[ms], false, false);
    }
  }

  const float* bias = (g == 0) ? b0 : (g == 1) ? b1 : (g == 2) ? b2 : b3;
  int n = n0 + (lid & 15);
  float bv = bias[n];
#pragma unroll
  for (int ms = 0; ms < 4; ++ms) {
    int mbase = mtg * 64 + ms * 16 + ((lid & 16) ? 8 : 0);
#pragma unroll
    for (int v = 0; v < 8; ++v) {
      int m = mbase + v;              // m = t*B + b
      int t = m >> 8;
      int b = m & 255;
      Xbuf16[(((size_t)t * 4 + g) * BB + b) * HH + n] =
          (_Float16)(acc[ms][v] + bv);
    }
  }
}

// ---------------------------------------------------------------------------
// Recurrence step (one launch per t). Block = one batch-tile (mt): 16 waves,
// each owning one h-tile (nt), all 4 gates. The shared 16x256 h panel (8 KB)
// is staged once in LDS via async copies. Weights stream from L2.
//   pre_g = h_{t-1} @ Wh_g^T + Xbuf16[t][g]   (bias already folded in)
//   c = sig(f)*c + sig(i)*tanh(cc);  h = sig(o)*tanh(c)
// h is double-buffered in f16 to avoid intra-launch races.
// ---------------------------------------------------------------------------
__global__ __launch_bounds__(512) void lstm_step(
    const _Float16* __restrict__ h16_in, _Float16* __restrict__ h16_out,
    const _Float16* __restrict__ Wh16, const _Float16* __restrict__ Xg16,
    float* __restrict__ cState, float* __restrict__ hState,
    float* __restrict__ hAll_t) {
  __shared__ __attribute__((aligned(16))) _Float16 Hsh[16 * HH];  // 8 KB

  int tid  = threadIdx.x;
  int lid  = tid & 31;
  int nt   = tid >> 5;        // wave index = h-tile, 0..15
  int mt   = blockIdx.x;      // batch-tile, 0..15
  int m0   = mt * 16, n0 = nt * 16;

  // Stage h panel: 16 rows x 512 B = 8 KB; 512 threads x one b128 each.
  {
    int r = tid >> 5;                       // 0..15
    int cb = (tid & 31) * 16;               // byte offset within row
    const char* src = (const char*)(h16_in + ((size_t)m0 + r) * HH) + cb;
    unsigned dst = (unsigned)(size_t)(&Hsh[0]) + (unsigned)(r * 512 + cb);
    async_copy_b128(dst, src);
  }
  wait_asynccnt0();
  __syncthreads();

  v8f zero = {0.f, 0.f, 0.f, 0.f, 0.f, 0.f, 0.f, 0.f};
  v8f acc[4] = {zero, zero, zero, zero};

#pragma unroll
  for (int k0 = 0; k0 < HH; k0 += 32) {
    v16h a = load_a_frag_lds(&Hsh[k0], HH, lid);
#pragma unroll
    for (int g = 0; g < 4; ++g) {
      v16h b = load_b_frag(Wh16 + (size_t)g * HH * HH + (size_t)n0 * HH + k0,
                           HH, lid);
      acc[g] = __builtin_amdgcn_wmma_f32_16x16x32_f16(
          false, a, false, b, (short)0, acc[g], false, false);
    }
  }

  int n = n0 + (lid & 15);
  int mbase = m0 + ((lid & 16) ? 8 : 0);
#pragma unroll
  for (int v = 0; v < 8; ++v) {
    int b   = mbase + v;
    int idx = b * HH + n;
    float ft = sigmoidf_(acc[0][v] + (float)Xg16[0 * BB * HH + idx]);
    float it = sigmoidf_(acc[1][v] + (float)Xg16[1 * BB * HH + idx]);
    float cc = tanhf(acc[2][v] + (float)Xg16[2 * BB * HH + idx]);
    float ot = sigmoidf_(acc[3][v] + (float)Xg16[3 * BB * HH + idx]);
    float cn = ft * cState[idx] + it * cc;
    cState[idx] = cn;
    float hn = ot * tanhf(cn);
    hState[idx]  = hn;
    h16_out[idx] = (_Float16)hn;
    hAll_t[idx]  = hn;
  }
}

__global__ __launch_bounds__(256) void finalize(const float* __restrict__ hState,
                                                const float* __restrict__ cState,
                                                float* __restrict__ out_h,
                                                float* __restrict__ out_c) {
  int i = blockIdx.x * 256 + threadIdx.x;
  if (i >= BB * HH) return;
  out_h[i] = hState[i];
  out_c[i] = cState[i];
}

// ---------------------------------------------------------------------------
extern "C" void kernel_launch(void* const* d_in, const int* in_sizes, int n_in,
                              void* d_out, int out_size, void* d_ws, size_t ws_size,
                              hipStream_t stream) {
  // Inputs (setup_inputs order): x,h,c, Wh,bh, Wx,bx, Wc,bc, Wo,bo
  const float* x  = (const float*)d_in[0];
  const float* h0 = (const float*)d_in[1];
  const float* c0 = (const float*)d_in[2];
  const float* Wf = (const float*)d_in[3];  const float* bf = (const float*)d_in[4];
  const float* Wi = (const float*)d_in[5];  const float* bi = (const float*)d_in[6];
  const float* Wc = (const float*)d_in[7];  const float* bc = (const float*)d_in[8];
  const float* Wo = (const float*)d_in[9];  const float* bo = (const float*)d_in[10];

  float* out = (float*)d_out;                         // [L*B*H | B*H | B*H]
  float* out_hfin = out + (size_t)LL * BB * HH;
  float* out_cfin = out_hfin + (size_t)BB * HH;

  // Workspace carve-up
  char* ws = (char*)d_ws;
  size_t off = 0;
  _Float16* x16    = (_Float16*)(ws + off); off += (size_t)LL * BB * DD * 2;   // 64 MB
  _Float16* Wh16   = (_Float16*)(ws + off); off += (size_t)4 * HH * HH * 2;    // 512 KB
  _Float16* Wx16   = (_Float16*)(ws + off); off += (size_t)4 * HH * DD * 2;    // 512 KB
  _Float16* h16A   = (_Float16*)(ws + off); off += (size_t)BB * HH * 2;        // 128 KB
  _Float16* h16B   = (_Float16*)(ws + off); off += (size_t)BB * HH * 2;        // 128 KB
  float* hState    = (float*)(ws + off);    off += (size_t)BB * HH * 4;        // 256 KB
  float* cState    = (float*)(ws + off);    off += (size_t)BB * HH * 4;        // 256 KB
  _Float16* Xbuf16 = (_Float16*)(ws + off); off += (size_t)LL * 4 * BB * HH * 2; // 256 MB

  // 1) Precision conversion / weight split
  convert_weights<<<(4 * HH * (HH + DD) + 255) / 256, 256, 0, stream>>>(
      Wf, Wi, Wc, Wo, Wh16, Wx16);
  int nx = LL * BB * DD;
  convert_x<<<(nx + 255) / 256, 256, 0, stream>>>(x, x16, nx);
  init_state<<<(BB * HH + 255) / 256, 256, 0, stream>>>(h0, c0, hState, cState,
                                                        h16A);

  // 2) Parallel x-contribution GEMM: 2048 m-groups x 8 gn-groups
  x_gemm<<<16384, 256, 0, stream>>>(x16, Wx16, bf, bi, bc, bo, Xbuf16);

  // 3) Sequential recurrence: 512 graph-captured step launches,
  //    16 blocks (one per batch-tile) x 16 waves (one per h-tile)
  for (int t = 0; t < LL; ++t) {
    const _Float16* hin  = (t & 1) ? h16B : h16A;
    _Float16*       hout = (t & 1) ? h16A : h16B;
    const _Float16* Xg = Xbuf16 + (size_t)t * 4 * BB * HH;
    float* hAll_t = out + (size_t)t * BB * HH;
    lstm_step<<<16, 512, 0, stream>>>(hin, hout, Wh16, Xg, cState, hState,
                                      hAll_t);
  }

  // 4) Final h, c
  finalize<<<(BB * HH + 255) / 256, 256, 0, stream>>>(hState, cState, out_hfin,
                                                      out_cfin);
}
